// VectorQuantizer_31044023615531
// MI455X (gfx1250) — compile-verified, matrix operand order checked by reading, not run on hardware
//
#include <hip/hip_runtime.h>

typedef __attribute__((ext_vector_type(16))) _Float16 v16h;
typedef __attribute__((ext_vector_type(8)))  _Float16 v8h;
typedef __attribute__((ext_vector_type(4)))  _Float16 v4h;
typedef __attribute__((ext_vector_type(8)))  float    v8f;

#define NPIX   262144   // B*H*W = 16*128*128
#define KCODES 1024
#define DDIM   64
#define NBLK   2048     // NPIX / 128 pixels-per-block

// workspace layout (float index):
// [0..1023]      a2[k]  = -2 * (w_in . emb_k)            (fp32, exact epilogue)
// [1024..2047]   g[k]   = ||emb_k||^2 - 2*(b_in . emb_k) (fp32)
// [2048..3071]   o[k]   = emb_k . w_out                  (fp32)
// [3072..5119]   v4h pk[k] = {a2_h, a2_l, g_h, g_l}      (8KB packed f16 splits)
// [5120..5123]   ww, wb, bb, (unused)
// [5124..7171]   per-block loss partial sums (2048)
#define WS_A2  0
#define WS_G   1024
#define WS_O   2048
#define WS_PK  3072
#define WS_SC  5120
#define WS_BS  5124

__global__ void vq_coeff_kernel(const float* __restrict__ w_in,
                                const float* __restrict__ b_in,
                                const float* __restrict__ emb,
                                const float* __restrict__ w_out,
                                float* __restrict__ ws) {
  int k = blockIdx.x * blockDim.x + threadIdx.x;
  if (k < KCODES) {
    const float* e = emb + k * DDIM;
    float a = 0.f, c = 0.f, o = 0.f, s = 0.f;
#pragma unroll 8
    for (int d = 0; d < DDIM; ++d) {
      float ev = e[d];
      a = fmaf(w_in[d], ev, a);
      c = fmaf(b_in[d], ev, c);
      o = fmaf(w_out[d], ev, o);
      s = fmaf(ev, ev, s);
    }
    float a2 = -2.0f * a;          // slope of score line
    float g  = s - 2.0f * c;       // intercept of score line
    ws[WS_A2 + k] = a2;
    ws[WS_G  + k] = g;
    ws[WS_O  + k] = o;
    _Float16 a2h = (_Float16)a2;
    _Float16 a2l = (_Float16)(a2 - (float)a2h);
    _Float16 gh  = (_Float16)g;
    _Float16 gl  = (_Float16)(g - (float)gh);
    v4h p; p.x = a2h; p.y = a2l; p.z = gh; p.w = gl;
    ((v4h*)(ws + WS_PK))[k] = p;
  }
  if (k == 0) {
    float ww = 0.f, wb = 0.f, bb = 0.f;
    for (int d = 0; d < DDIM; ++d) {
      ww = fmaf(w_in[d], w_in[d], ww);
      wb = fmaf(w_in[d], b_in[d], wb);
      bb = fmaf(b_in[d], b_in[d], bb);
    }
    ws[WS_SC + 0] = ww; ws[WS_SC + 1] = wb; ws[WS_SC + 2] = bb;
  }
}

__global__ __launch_bounds__(256) void vq_main_kernel(
    const float* __restrict__ x_in, const float* __restrict__ b_out,
    const float* __restrict__ wsr, float* __restrict__ ws,
    float* __restrict__ out) {
  // Pre-expanded, pre-masked A-fragments: one 16B entry per (tile, lane).
  // Lanes 0-15 (A rows, K-slots 0..7): {a2h, a2h, a2l, gh, gl, 0, 0, 0}
  // Lanes 16-31 (K-slots 8..15): all zero -> no per-tile cndmask needed.
  __shared__ v8h sA[64 * 32];    // 32 KB
  __shared__ float sSum[8];

  const float* A2 = wsr + WS_A2;
  const float* G  = wsr + WS_G;
  const float* O  = wsr + WS_O;
  const v4h*   PK = (const v4h*)(wsr + WS_PK);

  const int tid = threadIdx.x;
  for (int e = tid; e < 64 * 32; e += 256) {
    int l = e & 31;
    v8h a = {};
    if (l < 16) {
      v4h pk = PK[(e >> 5) * 16 + l];
      a[0] = pk.x; a[1] = pk.x; a[2] = pk.y; a[3] = pk.z; a[4] = pk.w;
    }
    sA[e] = a;
  }
  __syncthreads();

  const int wave = tid >> 5;
  const int lane = tid & 31;
  const bool lo16 = (lane < 16);
  const int  pix  = blockIdx.x * 128 + wave * 16 + (lane & 15);

  const float x = x_in[pix];
  // f16 split of x; B rows (K-slots): [x_h, x_l, x_h, 1, 1, 0...]
  _Float16 xh = (_Float16)x;
  _Float16 xl = (_Float16)(x - (float)xh);
  const _Float16 z16 = (_Float16)0.0f;
  _Float16 bxh = lo16 ? xh : z16;              // lanes>=16 carry K=8..15,24..31 -> zero
  _Float16 bxl = lo16 ? xl : z16;
  _Float16 bon = lo16 ? (_Float16)1.0f : z16;
  v16h bmat = {};
  bmat[0] = bxh; bmat[1] = bxl; bmat[2] = bxh; bmat[3] = bon; bmat[4] = bon;

  const v8h zer8 = {};
  const v8h* myA = sA + lane;

  // Two independent K-streams (tiles 0..31 and 32..63) so one stream's
  // min-reduce fills the other stream's WMMA->VALU hazard slots.
  float best0 = 3.0e38f, best1 = 3.0e38f;
  int tile0 = 0, tile1 = 32;
  for (int t = 0; t < 32; ++t) {
    v8h a0 = myA[t * 32];
    v8h a1 = myA[(t + 32) * 32];
    v16h A0 = __builtin_shufflevector(a0, zer8, 0, 1, 2, 3, 4, 5, 6, 7,
                                      8, 9, 10, 11, 12, 13, 14, 15);
    v16h A1 = __builtin_shufflevector(a1, zer8, 0, 1, 2, 3, 4, 5, 6, 7,
                                      8, 9, 10, 11, 12, 13, 14, 15);
    v8f cz0 = {}, cz1 = {};
    v8f d0 = __builtin_amdgcn_wmma_f32_16x16x32_f16(
        false, A0, false, bmat, (short)0, cz0, false, false);
    v8f d1 = __builtin_amdgcn_wmma_f32_16x16x32_f16(
        false, A1, false, bmat, (short)0, cz1, false, false);
    float t0 = fminf(fminf(fminf(d0[0], d0[1]), fminf(d0[2], d0[3])),
                     fminf(fminf(d0[4], d0[5]), fminf(d0[6], d0[7])));
    float t1 = fminf(fminf(fminf(d1[0], d1[1]), fminf(d1[2], d1[3])),
                     fminf(fminf(d1[4], d1[5]), fminf(d1[6], d1[7])));
    if (t0 < best0) { best0 = t0; tile0 = t; }
    if (t1 < best1) { best1 = t1; tile1 = t + 32; }
  }
  // merge streams; strict < keeps the lower code index on exact ties
  if (best1 < best0) { best0 = best1; tile0 = tile1; }

  // exact fp32 recompute over the winning tile's 8 codes for this lane-half
  const int off = (lane >> 4) << 3;            // 0 for lanes 0-15, 8 for 16-31
  const int cb  = tile0 * 16 + off;
  float bv = 3.0e38f;
  int bidx = cb;
#pragma unroll
  for (int r = 0; r < 8; ++r) {
    float sc = fmaf(A2[cb + r], x, G[cb + r]);
    if (sc < bv) { bv = sc; bidx = cb + r; }
  }
  // combine the two lane-halves covering the same pixel
  float ov = __shfl_xor(bv, 16, 32);
  int   oi = __shfl_xor(bidx, 16, 32);
  if (ov < bv || (ov == bv && oi < bidx)) { bv = ov; bidx = oi; }

  const float ww = wsr[WS_SC + 0], wb = wsr[WS_SC + 1], bb = wsr[WS_SC + 2];
  const float h2 = fmaf(x, fmaf(x, ww, 2.0f * wb), bb);    // ||h_n||^2
  const float dist = bv + h2;                               // ||q - h||^2

  if (lo16) out[pix] = O[bidx] + b_out[0];

  // deterministic loss partial: wave reduce (upper halves contribute 0)
  float contrib = lo16 ? dist : 0.0f;
#pragma unroll
  for (int s = 16; s > 0; s >>= 1) contrib += __shfl_xor(contrib, s, 32);
  if (lane == 0) sSum[wave] = contrib;
  __syncthreads();
  if (tid == 0) {
    float bsum = 0.f;
#pragma unroll
    for (int w = 0; w < 8; ++w) bsum += sSum[w];
    ws[WS_BS + blockIdx.x] = bsum;
  }
}

__global__ void vq_finalize_kernel(const float* __restrict__ ws,
                                   float* __restrict__ out) {
  if (threadIdx.x == 0 && blockIdx.x == 0) {
    float acc = 0.f;
    for (int i = 0; i < NBLK; ++i) acc += ws[WS_BS + i];
    // KLD * (CC + 1) * mean over N*D elements = 12.5 * acc / (N*D)
    out[NPIX] = 12.5f * acc * (1.0f / (float)(NPIX * DDIM));
  }
}

extern "C" void kernel_launch(void* const* d_in, const int* in_sizes, int n_in,
                              void* d_out, int out_size, void* d_ws, size_t ws_size,
                              hipStream_t stream) {
  const float* x     = (const float*)d_in[0];
  const float* w_in  = (const float*)d_in[1];
  const float* b_in  = (const float*)d_in[2];
  const float* emb   = (const float*)d_in[3];
  const float* w_out = (const float*)d_in[4];
  const float* b_out = (const float*)d_in[5];
  float* out = (float*)d_out;
  float* ws  = (float*)d_ws;

  vq_coeff_kernel<<<4, 256, 0, stream>>>(w_in, b_in, emb, w_out, ws);
  vq_main_kernel<<<NBLK, 256, 0, stream>>>(x, b_out, ws, ws, out);
  vq_finalize_kernel<<<1, 32, 0, stream>>>(ws, out);
}
